// Decoder_11965778887282
// MI455X (gfx1250) — matrix operand order repaired and checked by reading, not compiled, not run on hardware
//
#include <hip/hip_runtime.h>
#include <cstdint>
#include <cstddef>

// ---------------------------------------------------------------------------
// Decoder-with-attention forward loss for MI455X (gfx1250, wave32, WMMA).
// All four per-step matmuls run on v_wmma_f32_16x16x32_bf16 (f32 accum).
// Weights are packed f32->bf16 once per launch; activations are produced in
// bf16 directly by the epilogues of the producing kernels.
// ---------------------------------------------------------------------------

#define B_  32
#define H_  512
#define E_  512
#define V_  10000
#define TO_ 256
#define TI_ 1024

typedef __bf16 bf16_t;
typedef bf16_t v16bf __attribute__((ext_vector_type(16)));
typedef float  v8f   __attribute__((ext_vector_type(8)));
typedef unsigned int u32x4 __attribute__((ext_vector_type(4)));
typedef float  f32x4 __attribute__((ext_vector_type(4)));

union BF16Vec { u32x4 q[2]; v16bf v; };

__device__ __forceinline__ uint16_t f2bf(float f) {
  union { float f; uint32_t u; } x; x.f = f;
  uint32_t u = x.u;
  uint32_t r = u + 0x7FFFu + ((u >> 16) & 1u);   // round-to-nearest-even
  return (uint16_t)(r >> 16);
}
__device__ __forceinline__ float sigmoidf_(float x) {
  return 1.0f / (1.0f + expf(-x));
}

// ---------------------------------------------------------------------------
// Generic M=32 GEMM:  C[32,N] = act(A[32,K] * W[N,K]^T + bias)
// A, W in bf16 row-major; C f32; optional bf16 mirror of C.
// 4 waves per block; each wave owns one 16-wide N tile (32 rows).
// ---------------------------------------------------------------------------
__global__ __launch_bounds__(128)
void gemm32_bf16(const uint16_t* __restrict__ A,
                 const uint16_t* __restrict__ W,
                 const float* __restrict__ bias,
                 float* __restrict__ C,
                 uint16_t* __restrict__ Cbf,
                 int K, int N, int act)
{
  const int wave  = threadIdx.x >> 5;
  const int lane  = threadIdx.x & 31;
  const int ntile = blockIdx.x * 4 + wave;
  if (ntile * 16 >= N) return;                    // wave-uniform: EXEC stays all-1s

  const int n  = ntile * 16 + (lane & 15);
  const int hi = (lane & 16) ? 1 : 0;

  v8f acc0 = {}; v8f acc1 = {};
  const uint16_t* arow0 = A + (size_t)(lane & 15) * K + hi * 8;
  const uint16_t* arow1 = arow0 + (size_t)16 * K;
  const uint16_t* wrow  = W + (size_t)n * K + hi * 8;

  for (int kt = 0; kt < K; kt += 32) {
    BF16Vec a0, a1, b;
    a0.q[0] = *(const u32x4*)(arow0 + kt);        // K = base..base+7
    a0.q[1] = *(const u32x4*)(arow0 + kt + 16);   // K = base+16..base+23
    a1.q[0] = *(const u32x4*)(arow1 + kt);
    a1.q[1] = *(const u32x4*)(arow1 + kt + 16);
    b.q[0]  = *(const u32x4*)(wrow + kt);         // K = base..base+7
    b.q[1]  = *(const u32x4*)(wrow + kt + 16);    // K = base+16..base+23
    acc0 = __builtin_amdgcn_wmma_f32_16x16x32_bf16(false, a0.v, false, b.v,
                                                   (short)0, acc0, false, false);
    acc1 = __builtin_amdgcn_wmma_f32_16x16x32_bf16(false, a1.v, false, b.v,
                                                   (short)0, acc1, false, false);
  }

  const float bv = bias ? bias[n] : 0.0f;
  #pragma unroll
  for (int v = 0; v < 8; ++v) {
    const int m0 = v + hi * 8;                    // rows 0..15 (acc0), +16 (acc1)
    float x0 = acc0[v] + bv;
    float x1 = acc1[v] + bv;
    if (act == 1) { x0 = tanhf(x0); x1 = tanhf(x1); }
    C[(size_t)m0 * N + n]        = x0;
    C[(size_t)(m0 + 16) * N + n] = x1;
    if (Cbf) {
      Cbf[(size_t)m0 * N + n]        = f2bf(x0);
      Cbf[(size_t)(m0 + 16) * N + n] = f2bf(x1);
    }
  }
}

// ---------------------------------------------------------------------------
// Weight / bias packing (once per launch)
// ---------------------------------------------------------------------------
__global__ void pack_wc0(const float* __restrict__ Wih, const float* __restrict__ Whh,
                         uint16_t* __restrict__ Wc)   // [2048, 2048] = [Wih(1536) | Whh(512)]
{
  size_t idx = (size_t)blockIdx.x * blockDim.x + threadIdx.x;
  if (idx >= (size_t)2048 * 2048) return;
  int nrow = (int)(idx >> 11), k = (int)(idx & 2047);
  float v = (k < 1536) ? Wih[(size_t)nrow * 1536 + k]
                       : Whh[(size_t)nrow * 512 + (k - 1536)];
  Wc[idx] = f2bf(v);
}
__global__ void pack_wc1(const float* __restrict__ Wih, const float* __restrict__ Whh,
                         uint16_t* __restrict__ Wc)   // [2048, 1024] = [Wih(512) | Whh(512)]
{
  size_t idx = (size_t)blockIdx.x * blockDim.x + threadIdx.x;
  if (idx >= (size_t)2048 * 1024) return;
  int nrow = (int)(idx >> 10), k = (int)(idx & 1023);
  float v = (k < 512) ? Wih[(size_t)nrow * 512 + k]
                      : Whh[(size_t)nrow * 512 + (k - 512)];
  Wc[idx] = f2bf(v);
}
__global__ void pack_plain(const float* __restrict__ W, uint16_t* __restrict__ Wb, size_t n) {
  size_t idx = (size_t)blockIdx.x * blockDim.x + threadIdx.x;
  if (idx < n) Wb[idx] = f2bf(W[idx]);
}
__global__ void vec_add(const float* __restrict__ a, const float* __restrict__ b,
                        float* __restrict__ o, int n) {
  int i = blockIdx.x * blockDim.x + threadIdx.x;
  if (i < n) o[i] = a[i] + b[i];
}

// ---------------------------------------------------------------------------
// Embedding gather (bf16) into x0[:, 0:512].  ys_in[:,t] = t==0 ? SOS : tok[t-1]
// ---------------------------------------------------------------------------
__global__ __launch_bounds__(256)
void embed_kernel(const int* __restrict__ tok, const float* __restrict__ emb,
                  uint16_t* __restrict__ x0, int t)
{
  int idx = blockIdx.x * blockDim.x + threadIdx.x;   // B_*E_ = 16384
  int b = idx >> 9, j = idx & 511;
  int token = (t == 0) ? 1 : tok[b * TO_ + (t - 1)];
  x0[(size_t)b * 2048 + j] = f2bf(emb[(size_t)token * E_ + j]);
}

// ---------------------------------------------------------------------------
// LSTM pointwise: gates i,f,g,o from g[B,2048]; update h,c; emit bf16 h to two
// destinations (next-step GEMM input slots).
// ---------------------------------------------------------------------------
__global__ __launch_bounds__(256)
void lstm_pointwise(const float* __restrict__ g, float* __restrict__ h,
                    float* __restrict__ c,
                    uint16_t* __restrict__ dstA, int ldA,
                    uint16_t* __restrict__ dstB, int ldB)
{
  int idx = blockIdx.x * blockDim.x + threadIdx.x;   // B_*H_ = 16384
  int b = idx >> 9, j = idx & 511;
  const float* gr = g + (size_t)b * 2048;
  float ig = sigmoidf_(gr[j]);
  float fg = sigmoidf_(gr[512 + j]);
  float gg = tanhf(gr[1024 + j]);
  float og = sigmoidf_(gr[1536 + j]);
  float cn = fg * c[idx] + ig * gg;
  float hn = og * tanhf(cn);
  c[idx] = cn;
  h[idx] = hn;
  uint16_t hb = f2bf(hn);
  dstA[(size_t)b * ldA + j] = hb;
  dstB[(size_t)b * ldB + j] = hb;
}

// ---------------------------------------------------------------------------
// Attention scores: one wave per (b, ti) dot product of length 512 (coalesced).
// ---------------------------------------------------------------------------
__global__ __launch_bounds__(256)
void attn_scores(const float* __restrict__ h1, const float* __restrict__ enc,
                 float* __restrict__ scores)
{
  int gw   = (blockIdx.x * blockDim.x + threadIdx.x) >> 5;  // 0 .. B_*TI_-1
  int lane = threadIdx.x & 31;
  int b = gw >> 10, ti = gw & 1023;
  const float* e = enc + ((size_t)b * TI_ + ti) * H_;
  const float* h = h1 + (size_t)b * H_;
  float s = 0.f;
  #pragma unroll
  for (int j = lane * 4; j < H_; j += 128) {
    f32x4 ev = *(const f32x4*)(e + j);
    f32x4 hv = *(const f32x4*)(h + j);
    s += ev.x * hv.x + ev.y * hv.y + ev.z * hv.z + ev.w * hv.w;
  }
  #pragma unroll
  for (int off = 16; off > 0; off >>= 1) s += __shfl_xor(s, off, 32);
  if (lane == 0) scores[(size_t)b * TI_ + ti] = s;
}

__global__ __launch_bounds__(256)
void attn_softmax(float* __restrict__ scores)
{
  int b = blockIdx.x, tid = threadIdx.x;
  float* s = scores + (size_t)b * TI_;
  __shared__ float red[256];
  float m = -1e30f;
  for (int i = tid; i < TI_; i += 256) m = fmaxf(m, s[i]);
  red[tid] = m; __syncthreads();
  for (int st = 128; st > 0; st >>= 1) { if (tid < st) red[tid] = fmaxf(red[tid], red[tid + st]); __syncthreads(); }
  m = red[0]; __syncthreads();
  float sum = 0.f;
  for (int i = tid; i < TI_; i += 256) { float e = expf(s[i] - m); s[i] = e; sum += e; }
  red[tid] = sum; __syncthreads();
  for (int st = 128; st > 0; st >>= 1) { if (tid < st) red[tid] += red[tid + st]; __syncthreads(); }
  float inv = 1.f / red[0];
  for (int i = tid; i < TI_; i += 256) s[i] *= inv;
}

// context[b,h] = sum_t w[b,t]*enc[b,t,h]; bf16 into x0[:,512:1024] & mlp[:,512:1024]
__global__ __launch_bounds__(512)
void attn_context(const float* __restrict__ att_w, const float* __restrict__ enc,
                  uint16_t* __restrict__ x0, uint16_t* __restrict__ mlp)
{
  int b = blockIdx.x, h = threadIdx.x;               // 512 threads
  __shared__ float w[TI_];
  for (int i = h; i < TI_; i += 512) w[i] = att_w[(size_t)b * TI_ + i];
  __syncthreads();
  const float* e = enc + (size_t)b * TI_ * H_ + h;
  float acc = 0.f;
  for (int t = 0; t < TI_; ++t) acc += w[t] * e[(size_t)t * H_];
  uint16_t v = f2bf(acc);
  x0 [(size_t)b * 2048 + 512 + h] = v;
  mlp[(size_t)b * 1024 + 512 + h] = v;
}

// ---------------------------------------------------------------------------
// Masked NLL over one step's logits [32, V]; atomically accumulate sum & count.
// ---------------------------------------------------------------------------
__global__ __launch_bounds__(256)
void loss_kernel(const float* __restrict__ logits, const int* __restrict__ tok,
                 int t, float* __restrict__ acc)
{
  int b = blockIdx.x, tid = threadIdx.x;
  const float* lr = logits + (size_t)b * V_;
  int label = (t < TO_) ? tok[b * TO_ + t] : 2;      // ys_out; EOS at final step
  __shared__ float red[256];
  float m = -1e30f;
  for (int i = tid; i < V_; i += 256) m = fmaxf(m, lr[i]);
  red[tid] = m; __syncthreads();
  for (int st = 128; st > 0; st >>= 1) { if (tid < st) red[tid] = fmaxf(red[tid], red[tid + st]); __syncthreads(); }
  m = red[0]; __syncthreads();
  float sum = 0.f;
  for (int i = tid; i < V_; i += 256) sum += expf(lr[i] - m);
  red[tid] = sum; __syncthreads();
  for (int st = 128; st > 0; st >>= 1) { if (tid < st) red[tid] += red[tid + st]; __syncthreads(); }
  if (tid == 0 && label != 0) {
    float logp = lr[label] - m - logf(red[0]);
    atomicAdd(&acc[0], -logp);
    atomicAdd(&acc[1], 1.0f);
  }
}

__global__ void finalize_kernel(const float* __restrict__ acc, float* __restrict__ out) {
  out[0] = acc[0] / fmaxf(acc[1], 1.0f);
}

// ---------------------------------------------------------------------------
extern "C" void kernel_launch(void* const* d_in, const int* in_sizes, int n_in,
                              void* d_out, int out_size, void* d_ws, size_t ws_size,
                              hipStream_t stream)
{
  const int*   tok  = (const int*)  d_in[0];
  const float* enc  = (const float*)d_in[1];
  const float* emb  = (const float*)d_in[2];
  const float* Wih0 = (const float*)d_in[3];
  const float* Whh0 = (const float*)d_in[4];
  const float* bih0 = (const float*)d_in[5];
  const float* bhh0 = (const float*)d_in[6];
  const float* Wih1 = (const float*)d_in[7];
  const float* Whh1 = (const float*)d_in[8];
  const float* bih1 = (const float*)d_in[9];
  const float* bhh1 = (const float*)d_in[10];
  const float* W1   = (const float*)d_in[11];
  const float* b1   = (const float*)d_in[12];
  const float* W2   = (const float*)d_in[13];
  const float* b2   = (const float*)d_in[14];

  char* ws = (char*)d_ws;
  size_t off = 0;
  auto alloc = [&](size_t bytes) -> char* {
    char* p = ws + off;
    off = (off + bytes + 255) & ~(size_t)255;
    return p;
  };

  // bf16 packed weights (no init needed; rewritten every launch)
  uint16_t* Wc0b = (uint16_t*)alloc((size_t)2048 * 2048 * 2);
  uint16_t* Wc1b = (uint16_t*)alloc((size_t)2048 * 1024 * 2);
  uint16_t* W1b  = (uint16_t*)alloc((size_t)H_ * 1024 * 2);
  uint16_t* W2b  = (uint16_t*)alloc((size_t)V_ * H_ * 2);
  float*    b0c  = (float*)   alloc(2048 * 4);
  float*    b1c  = (float*)   alloc(2048 * 4);
  // scratch written before read every step
  float*    g0     = (float*)   alloc((size_t)B_ * 2048 * 4);
  float*    g1     = (float*)   alloc((size_t)B_ * 2048 * 4);
  float*    hidden = (float*)   alloc((size_t)B_ * H_ * 4);
  uint16_t* hid_bf = (uint16_t*)alloc((size_t)B_ * H_ * 2);
  float*    scores = (float*)   alloc((size_t)B_ * TI_ * 4);
  float*    logits = (float*)   alloc((size_t)B_ * V_ * 4);
  // carried state (zero-initialized region, contiguous)
  size_t zoff = off;
  float*    h0  = (float*)   alloc((size_t)B_ * H_ * 4);
  float*    c0  = (float*)   alloc((size_t)B_ * H_ * 4);
  float*    h1  = (float*)   alloc((size_t)B_ * H_ * 4);
  float*    c1  = (float*)   alloc((size_t)B_ * H_ * 4);
  uint16_t* x0  = (uint16_t*)alloc((size_t)B_ * 2048 * 2);  // [emb | att_c | h0]
  uint16_t* x1  = (uint16_t*)alloc((size_t)B_ * 1024 * 2);  // [h0 | h1]
  uint16_t* mlp = (uint16_t*)alloc((size_t)B_ * 1024 * 2);  // [h1 | att_c]
  float*    acc = (float*)   alloc(2 * 4);
  size_t zbytes = off - zoff;

  // -- one-time packing + state init ---------------------------------------
  pack_wc0<<<(2048 * 2048 + 255) / 256, 256, 0, stream>>>(Wih0, Whh0, Wc0b);
  pack_wc1<<<(2048 * 1024 + 255) / 256, 256, 0, stream>>>(Wih1, Whh1, Wc1b);
  pack_plain<<<((size_t)H_ * 1024 + 255) / 256, 256, 0, stream>>>(W1, W1b, (size_t)H_ * 1024);
  pack_plain<<<((size_t)V_ * H_ + 255) / 256, 256, 0, stream>>>(W2, W2b, (size_t)V_ * H_);
  vec_add<<<8, 256, 0, stream>>>(bih0, bhh0, b0c, 2048);
  vec_add<<<8, 256, 0, stream>>>(bih1, bhh1, b1c, 2048);
  hipMemsetAsync(ws + zoff, 0, zbytes, stream);

  // -- sequential decode loop (257 steps) ----------------------------------
  for (int t = 0; t <= TO_; ++t) {
    embed_kernel<<<64, 256, 0, stream>>>(tok, emb, x0, t);
    // LSTM0: g0 = [emb|att_c|h0] @ [Wih0;Whh0]^T + (bih0+bhh0)   K=2048, N=2048
    gemm32_bf16<<<32, 128, 0, stream>>>(x0, Wc0b, b0c, g0, nullptr, 2048, 2048, 0);
    lstm_pointwise<<<64, 256, 0, stream>>>(g0, h0, c0, x1, 1024, x0 + 1536, 2048);
    // LSTM1: g1 = [h0|h1] @ [Wih1;Whh1]^T + (bih1+bhh1)          K=1024, N=2048
    gemm32_bf16<<<32, 128, 0, stream>>>(x1, Wc1b, b1c, g1, nullptr, 1024, 2048, 0);
    lstm_pointwise<<<64, 256, 0, stream>>>(g1, h1, c1, x1 + 512, 1024, mlp, 1024);
    // attention
    attn_scores<<<4096, 256, 0, stream>>>(h1, enc, scores);
    attn_softmax<<<32, 256, 0, stream>>>(scores);
    attn_context<<<32, 512, 0, stream>>>(scores, enc, x0, mlp);
    // MLP: hidden = tanh([h1|att_c] @ W1^T + b1)                 K=1024, N=512
    gemm32_bf16<<<8, 128, 0, stream>>>(mlp, W1b, b1, hidden, hid_bf, 1024, H_, 1);
    // logits = hidden @ W2^T + b2                                K=512, N=10000
    gemm32_bf16<<<157, 128, 0, stream>>>(hid_bf, W2b, b2, logits, nullptr, H_, V_, 0);
    loss_kernel<<<32, 256, 0, stream>>>(logits, tok, t, acc);
  }
  finalize_kernel<<<1, 1, 0, stream>>>(acc, (float*)d_out);
}
